// TemperGraph_3917010174337
// MI455X (gfx1250) — matrix operand order repaired and checked by reading, not compiled, not run on hardware
//
#include <hip/hip_runtime.h>
#include <math.h>

#define N_TOK   65536
#define D_INN   128
#define HID     64
#define T_TEMP  12
#define O_OPS   3
#define N_HOPS  4

typedef __attribute__((ext_vector_type(16))) __bf16 v16bf;
typedef __attribute__((ext_vector_type(8)))  float  v8f;

union FragAB {
  v16bf v;
  unsigned short u[16];
  unsigned w32[8];
  uint4 q[2];
};

__device__ __forceinline__ unsigned short f2bf(float f) {   // scalar RNE (prep only)
  unsigned u = __float_as_uint(f);
  unsigned r = u + 0x7FFFu + ((u >> 16) & 1u);
  return (unsigned short)(r >> 16);
}

// Packed f32x2 -> bf16x2 (RNE) in one VALU op.
__device__ __forceinline__ unsigned pk_bf16(float a, float b) {
#if __has_builtin(__builtin_amdgcn_cvt_pk_bf16_f32)
  typedef __attribute__((ext_vector_type(2))) __bf16 v2bf;
  union { v2bf v; unsigned u; } cv;
  cv.v = __builtin_amdgcn_cvt_pk_bf16_f32(a, b);
  return cv.u;
#else
  unsigned r;
  asm("v_cvt_pk_bf16_f32 %0, %1, %2" : "=v"(r) : "v"(a), "v"(b));
  return r;
#endif
}

__device__ __forceinline__ unsigned short cvt_bf16(float a) {
  return (unsigned short)pk_bf16(a, a);
}

// Threefry-2x32, 20 rounds (JAX PRNG core).
__device__ __forceinline__ void tf2x32(unsigned k0, unsigned k1,
                                       unsigned x0, unsigned x1,
                                       unsigned* o0, unsigned* o1) {
  const unsigned rotA[4] = {13u, 15u, 26u, 6u};
  const unsigned rotB[4] = {17u, 29u, 16u, 24u};
  unsigned ks[3] = {k0, k1, k0 ^ k1 ^ 0x1BD11BDAu};
  x0 += ks[0]; x1 += ks[1];
#pragma unroll
  for (int g = 0; g < 5; ++g) {
    const unsigned* rot = (g & 1) ? rotB : rotA;
#pragma unroll
    for (int r = 0; r < 4; ++r) {
      x0 += x1;
      x1 = (x1 << rot[r]) | (x1 >> (32u - rot[r]));
      x1 ^= x0;
    }
    x0 += ks[(g + 1) % 3];
    x1 += ks[(g + 2) % 3] + (unsigned)(g + 1);
  }
  *o0 = x0; *o1 = x1;
}

__device__ __forceinline__ float u01(unsigned b) {
  return ((float)(b >> 8) + 0.5f) * (1.0f / 16777216.0f);
}

// A fragment from f32 memory (proj only): pairs (k,k+1) are in-lane -> pk_bf16.
__device__ __forceinline__ FragAB makeA_f32(const float* row, int kb, int hi) {
  int k0 = 32 * kb + 8 * hi;
  float4 a0 = *(const float4*)(row + k0);
  float4 a1 = *(const float4*)(row + k0 + 4);
  float4 a2 = *(const float4*)(row + k0 + 16);
  float4 a3 = *(const float4*)(row + k0 + 20);
  FragAB f;
  f.w32[0] = pk_bf16(a0.x, a0.y); f.w32[1] = pk_bf16(a0.z, a0.w);
  f.w32[2] = pk_bf16(a1.x, a1.y); f.w32[3] = pk_bf16(a1.z, a1.w);
  f.w32[4] = pk_bf16(a2.x, a2.y); f.w32[5] = pk_bf16(a2.z, a2.w);
  f.w32[6] = pk_bf16(a3.x, a3.y); f.w32[7] = pk_bf16(a3.z, a3.w);
  return f;
}

// A fragment from bf16 memory (LDS or global): two 16B runs, zero VALU.
__device__ __forceinline__ FragAB makeA_bf16(const unsigned short* row, int kb, int hi) {
  int k0 = 32 * kb + 8 * hi;
  FragAB f;
  f.q[0] = *(const uint4*)(row + k0);
  f.q[1] = *(const uint4*)(row + k0 + 16);
  return f;
}

// B fragment (32x16 bf16, N=lane&15): two 8-element K runs at 32*kb+8*hi and +16
// from pre-transposed bf16 weights wt[n][K].
__device__ __forceinline__ FragAB loadB(const unsigned short* wt, int K, int n, int kb, int hi) {
  const unsigned short* p = wt + n * K + 32 * kb + 8 * hi;
  FragAB f;
  f.q[0] = *(const uint4*)p;
  f.q[1] = *(const uint4*)(p + 16);
  return f;
}

__device__ __forceinline__ v8f wmma_bf16(FragAB a, FragAB b, v8f c) {
  return __builtin_amdgcn_wmma_f32_16x16x32_bf16(false, a.v, false, b.v,
                                                 (short)0, c, false, false);
}

// ---------------- weight prep: transpose + f32->bf16 ----------------
__global__ void prep_weights(const float* projW, const float* W1, const float* W2,
                             const float* RW1, const float* RW2, const float* predW,
                             unsigned short* projWt, unsigned short* W1t,
                             unsigned short* W2t, unsigned short* RW1t,
                             unsigned short* RW2t, unsigned short* predWt) {
  const int total = 8192 + 147456 + 147456 + 4096 + 1024 + 8192;
  for (int i = blockIdx.x * blockDim.x + threadIdx.x; i < total;
       i += gridDim.x * blockDim.x) {
    int idx = i;
    if (idx < 8192) {                       // projWt[n][128] from projW[128][64]
      int n = idx >> 7, k = idx & 127;
      projWt[idx] = f2bf(projW[k * 64 + n]);
      continue;
    }
    idx -= 8192;
    if (idx < 147456) {                     // W1t[g][n][64] from W1[g][64][64]
      int g = idx >> 12, r = idx & 4095, n = r >> 6, k = r & 63;
      W1t[idx] = f2bf(W1[g * 4096 + k * 64 + n]);
      continue;
    }
    idx -= 147456;
    if (idx < 147456) {
      int g = idx >> 12, r = idx & 4095, n = r >> 6, k = r & 63;
      W2t[idx] = f2bf(W2[g * 4096 + k * 64 + n]);
      continue;
    }
    idx -= 147456;
    if (idx < 4096) {                       // RW1t[n][64]
      int n = idx >> 6, k = idx & 63;
      RW1t[idx] = f2bf(RW1[k * 64 + n]);
      continue;
    }
    idx -= 4096;
    if (idx < 1024) {                       // RW2t[16][64], pad cols 13..15 = 0
      int n = idx >> 6, k = idx & 63;
      RW2t[idx] = (n < T_TEMP + 1) ? f2bf(RW2[k * (T_TEMP + 1) + n]) : (unsigned short)0;
      continue;
    }
    idx -= 1024;
    {                                       // predWt[n][64] from predW[64][128]
      int n = idx >> 6, k = idx & 63;
      predWt[idx] = f2bf(predW[k * 128 + n]);
    }
  }
}

// ---------------- projection + state init (states stored bf16) ----------------
__global__ __launch_bounds__(32) void proj_kernel(
    const float* x, const unsigned short* projWt, const float* projB,
    const int* initT, unsigned short* states, int* tempers, int* done) {
  int tile = blockIdx.x;
  int lane = threadIdx.x, m = lane & 15, hi = lane >> 4;
  int row = tile * 16 + m;
  FragAB A[4];
#pragma unroll
  for (int kb = 0; kb < 4; ++kb) A[kb] = makeA_f32(x + (long)row * D_INN, kb, hi);
#pragma unroll
  for (int j = 0; j < 4; ++j) {
    FragAB B[4];
#pragma unroll
    for (int kb = 0; kb < 4; ++kb) B[kb] = loadB(projWt, D_INN, 16 * j + m, kb, hi);
    v8f c = {};
#pragma unroll
    for (int kb = 0; kb < 4; ++kb) c = wmma_bf16(A[kb], B[kb], c);
    float bias = projB[16 * j + m];
#pragma unroll
    for (int v = 0; v < 8; ++v)
      states[(long)(tile * 16 + v + 8 * hi) * HID + 16 * j + m] = cvt_bf16(c[v] + bias);
  }
  if (lane < 16) { tempers[row] = initT[row]; done[row] = 0; }
}

// ---------------- bucketing ----------------
__global__ void zero_ctrl(int* ctrl) {
  if (threadIdx.x < 64) ctrl[threadIdx.x] = 0;
}

__global__ void hist_kernel(const int* tempers, const int* done, int* ctrl) {
  __shared__ int h[T_TEMP];
  if (threadIdx.x < T_TEMP) h[threadIdx.x] = 0;
  __syncthreads();
  int i = blockIdx.x * blockDim.x + threadIdx.x;
  if (i < N_TOK && !done[i]) atomicAdd(&h[tempers[i]], 1);
  __syncthreads();
  if (threadIdx.x < T_TEMP && h[threadIdx.x] > 0)
    atomicAdd(&ctrl[threadIdx.x], h[threadIdx.x]);
}

// ctrl[0..11]=counts, ctrl[12..23]=fill, ctrl[24..36]=tileBase (ctrl[36]=totalTiles)
__global__ void offsets_kernel(int* ctrl, int* bucket) {
  if (threadIdx.x == 0) {
    int tb = 0;
    for (int t = 0; t < T_TEMP; ++t) {
      ctrl[24 + t] = tb;
      int cnt = ctrl[t];
      int tiles = (cnt + 15) >> 4;
      for (int s = cnt; s < tiles * 16; ++s) bucket[tb * 16 + s] = -1;
      ctrl[12 + t] = 0;
      tb += tiles;
    }
    ctrl[24 + T_TEMP] = tb;
  }
}

__global__ void scatter_kernel(const int* tempers, const int* done, int* ctrl, int* bucket) {
  __shared__ int lh[T_TEMP];
  __shared__ int lbase[T_TEMP];
  if (threadIdx.x < T_TEMP) lh[threadIdx.x] = 0;
  __syncthreads();
  int i = blockIdx.x * blockDim.x + threadIdx.x;
  int t = -1, lp = 0;
  if (i < N_TOK && !done[i]) {
    t = tempers[i];
    lp = atomicAdd(&lh[t], 1);
  }
  __syncthreads();
  if (threadIdx.x < T_TEMP && lh[threadIdx.x] > 0)
    lbase[threadIdx.x] = atomicAdd(&ctrl[12 + threadIdx.x], lh[threadIdx.x]);
  __syncthreads();
  if (t >= 0)
    bucket[ctrl[24 + t] * 16 + lbase[t] + lp] = i;
}

// ---------------- per-hop temper MLP + routing + sampling ----------------
__global__ __launch_bounds__(32) void mlp_kernel(
    unsigned short* states, int* tempers, int* done,
    const int* ctrl, const int* bucket,
    const unsigned short* W1t, const unsigned short* W2t,
    const float* b1, const float* b2,
    const unsigned short* RW1t, const float* rb1,
    const unsigned short* RW2t, const float* rb2, int hop) {
  __shared__ __align__(16) unsigned short bufS[16][64];  // async-staged state tile
  __shared__ __align__(16) unsigned short bufA[16][72];  // h1 / r1 (bf16)
  __shared__ __align__(16) unsigned short bufB[16][72];  // out (bf16)
  __shared__ float logitsBuf[16][16];
  __shared__ int rids[16];

  int w = blockIdx.x;
  if (w >= ctrl[24 + T_TEMP]) return;
  int lane = threadIdx.x, m = lane & 15, hi = lane >> 4;

  int t = 0;
#pragma unroll
  for (int q = 1; q < T_TEMP; ++q)
    if (w >= ctrl[24 + q]) t = q;

  if (lane < 16) rids[lane] = bucket[w * 16 + lane];
  __syncthreads();
  int myRow = rids[m];

  // Async-stage the 16x64 bf16 state tile (2KB) into LDS, overlapped with RNG.
#pragma unroll
  for (int it = 0; it < 4; ++it) {
    int chunk = lane + 32 * it;                 // 0..127 : 16 rows x 8 chunks
    int mm = chunk >> 3, qq = chunk & 7;
    int r = rids[mm];
    int rr = r < 0 ? 0 : r;
    unsigned long long ga =
        (unsigned long long)(states + (long)rr * HID + 8 * qq);
    unsigned lo = (unsigned)(unsigned long long)(&bufS[mm][8 * qq]);
    asm volatile("global_load_async_to_lds_b128 %0, %1, off"
                 :: "v"(lo), "v"(ga) : "memory");
  }

  // RNG keys: rkey=(0,42); hkey=fold(rkey,hop); tkey=fold(hkey,t); skey=fold(hkey,10000)
  unsigned hk0, hk1, tk0, tk1, sk0, sk1;
  tf2x32(0u, 42u, 0u, (unsigned)hop, &hk0, &hk1);
  tf2x32(hk0, hk1, 0u, (unsigned)t, &tk0, &tk1);
  tf2x32(hk0, hk1, 0u, 10000u, &sk0, &sk1);

  float wn[O_OPS];
  {
#pragma unroll
    for (int o = 0; o < O_OPS; ++o) {
      unsigned r0, r1;
      tf2x32(tk0, tk1, (unsigned)o, 0u, &r0, &r1);
      wn[o] = sqrtf(-2.0f * logf(u01(r0))) * cosf(6.28318530718f * u01(r1));
    }
    float mx = fmaxf(wn[0], fmaxf(wn[1], wn[2]));
    float e0 = expf(wn[0] - mx), e1 = expf(wn[1] - mx), e2 = expf(wn[2] - mx);
    float inv = 1.0f / (e0 + e1 + e2);
    wn[0] = e0 * inv; wn[1] = e1 * inv; wn[2] = e2 * inv;
  }

  asm volatile("s_wait_asynccnt 0" ::: "memory");
  __syncthreads();

  FragAB As[2];
  As[0] = makeA_bf16(&bufS[m][0], 0, hi);
  As[1] = makeA_bf16(&bufS[m][0], 1, hi);

  v8f outAcc[4] = {};
#pragma unroll
  for (int o = 0; o < O_OPS; ++o) {
    int g = t * O_OPS + o;
    const unsigned short* w1 = W1t + g * HID * HID;
    const unsigned short* w2 = W2t + g * HID * HID;
    const float* bb1 = b1 + g * HID;
    const float* bb2 = b2 + g * HID;

    // layer 1: preload all 8 B fragments, then 8 back-to-back WMMAs
    FragAB B1[8];
#pragma unroll
    for (int j = 0; j < 4; ++j) {
      B1[2 * j]     = loadB(w1, HID, 16 * j + m, 0, hi);
      B1[2 * j + 1] = loadB(w1, HID, 16 * j + m, 1, hi);
    }
    __syncthreads();
#pragma unroll
    for (int j = 0; j < 4; ++j) {
      v8f c = {};
      c = wmma_bf16(As[0], B1[2 * j], c);
      c = wmma_bf16(As[1], B1[2 * j + 1], c);
      float bias = bb1[16 * j + m];
#pragma unroll
      for (int v = 0; v < 8; ++v)
        bufA[v + 8 * hi][16 * j + m] = cvt_bf16(fmaxf(c[v] + bias, 0.0f));
    }
    __syncthreads();
    FragAB Ah0 = makeA_bf16(&bufA[m][0], 0, hi);
    FragAB Ah1 = makeA_bf16(&bufA[m][0], 1, hi);
    FragAB B2[8];
#pragma unroll
    for (int j = 0; j < 4; ++j) {
      B2[2 * j]     = loadB(w2, HID, 16 * j + m, 0, hi);
      B2[2 * j + 1] = loadB(w2, HID, 16 * j + m, 1, hi);
    }
#pragma unroll
    for (int j = 0; j < 4; ++j) {
      v8f c = {};
      c = wmma_bf16(Ah0, B2[2 * j], c);
      c = wmma_bf16(Ah1, B2[2 * j + 1], c);
      float bias = bb2[16 * j + m];
#pragma unroll
      for (int v = 0; v < 8; ++v)
        outAcc[j][v] += wn[o] * fmaxf(c[v] + bias, 0.0f);
    }
  }

  __syncthreads();
#pragma unroll
  for (int j = 0; j < 4; ++j)
#pragma unroll
    for (int v = 0; v < 8; ++v)
      bufB[v + 8 * hi][16 * j + m] = cvt_bf16(outAcc[j][v]);
  __syncthreads();

  // routing layer 1
  FragAB Ao0 = makeA_bf16(&bufB[m][0], 0, hi);
  FragAB Ao1 = makeA_bf16(&bufB[m][0], 1, hi);
  {
    FragAB BR[8];
#pragma unroll
    for (int j = 0; j < 4; ++j) {
      BR[2 * j]     = loadB(RW1t, HID, 16 * j + m, 0, hi);
      BR[2 * j + 1] = loadB(RW1t, HID, 16 * j + m, 1, hi);
    }
#pragma unroll
    for (int j = 0; j < 4; ++j) {
      v8f c = {};
      c = wmma_bf16(Ao0, BR[2 * j], c);
      c = wmma_bf16(Ao1, BR[2 * j + 1], c);
      float bias = rb1[16 * j + m];
#pragma unroll
      for (int v = 0; v < 8; ++v)
        bufA[v + 8 * hi][16 * j + m] = cvt_bf16(fmaxf(c[v] + bias, 0.0f));
    }
  }
  __syncthreads();
  // routing layer 2 -> logits (13 classes, padded to 16)
  FragAB Ar0 = makeA_bf16(&bufA[m][0], 0, hi);
  FragAB Ar1 = makeA_bf16(&bufA[m][0], 1, hi);
  {
    FragAB Bq0 = loadB(RW2t, HID, m, 0, hi);
    FragAB Bq1 = loadB(RW2t, HID, m, 1, hi);
    v8f c = {};
    c = wmma_bf16(Ar0, Bq0, c);
    c = wmma_bf16(Ar1, Bq1, c);
    float bias = (m < T_TEMP + 1) ? rb2[m] : 0.0f;
#pragma unroll
    for (int v = 0; v < 8; ++v)
      logitsBuf[v + 8 * hi][m] = c[v] + bias;
  }
  __syncthreads();

  // Gumbel-max categorical sampling (classes 0..T), temper/done update
  if (lane < 16 && myRow >= 0) {
    int best = 0;
    float bestv = -1e38f;
    for (int cidx = 0; cidx <= T_TEMP; ++cidx) {
      unsigned g0, g1;
      tf2x32(sk0, sk1, (unsigned)myRow, (unsigned)cidx, &g0, &g1);
      float gum = -logf(-logf(u01(g0)));
      float val = logitsBuf[m][cidx] + gum;
      if (val > bestv) { bestv = val; best = cidx; }
    }
    tempers[myRow] = (best >= T_TEMP) ? (T_TEMP - 1) : best;
    if (best == T_TEMP) done[myRow] = 1;
  }
  // states <- out for all (active) tokens in this tile (16B chunks)
#pragma unroll
  for (int it = 0; it < 4; ++it) {
    int chunk = lane + 32 * it;                 // 0..127 : 16 rows x 8 chunks
    int mm = chunk >> 3, qq = chunk & 7;
    int r = rids[mm];
    if (r >= 0)
      *(uint4*)(states + (long)r * HID + 8 * qq) = *(const uint4*)(&bufB[mm][8 * qq]);
  }
}

// ---------------- pred = states @ pred_W + b ----------------
__global__ __launch_bounds__(32) void pred_kernel(
    const unsigned short* states, const unsigned short* predWt,
    const float* predB, float* pred) {
  int tile = blockIdx.x;
  int lane = threadIdx.x, m = lane & 15, hi = lane >> 4;
  int row = tile * 16 + m;
  FragAB A0 = makeA_bf16(states + (long)row * HID, 0, hi);
  FragAB A1 = makeA_bf16(states + (long)row * HID, 1, hi);
#pragma unroll
  for (int half = 0; half < 2; ++half) {
    FragAB B[8];
#pragma unroll
    for (int jj = 0; jj < 4; ++jj) {
      int j = half * 4 + jj;
      B[2 * jj]     = loadB(predWt, HID, 16 * j + m, 0, hi);
      B[2 * jj + 1] = loadB(predWt, HID, 16 * j + m, 1, hi);
    }
#pragma unroll
    for (int jj = 0; jj < 4; ++jj) {
      int j = half * 4 + jj;
      v8f c = {};
      c = wmma_bf16(A0, B[2 * jj], c);
      c = wmma_bf16(A1, B[2 * jj + 1], c);
      float bias = predB[16 * j + m];
#pragma unroll
      for (int v = 0; v < 8; ++v)
        pred[(long)(tile * 16 + v + 8 * hi) * D_INN + 16 * j + m] = c[v] + bias;
    }
  }
}

// ---------------- err = mean((pred-x)^2, -1) : one wave per token ----------------
__global__ void err_kernel(const float* pred, const float* x, float* err) {
  int tok = (blockIdx.x * blockDim.x + threadIdx.x) >> 5;
  int lane = threadIdx.x & 31;
  if (tok >= N_TOK) return;
  float4 pv = *(const float4*)(pred + (long)tok * D_INN + lane * 4);
  float4 xv = *(const float4*)(x + (long)tok * D_INN + lane * 4);
  float dx = pv.x - xv.x, dy = pv.y - xv.y, dz = pv.z - xv.z, dw = pv.w - xv.w;
  float s = dx * dx + dy * dy + dz * dz + dw * dw;
#pragma unroll
  for (int off = 16; off > 0; off >>= 1) s += __shfl_xor(s, off, 32);
  if (lane == 0) err[tok] = s * (1.0f / (float)D_INN);
}

extern "C" void kernel_launch(void* const* d_in, const int* in_sizes, int n_in,
                              void* d_out, int out_size, void* d_ws, size_t ws_size,
                              hipStream_t stream) {
  const float* x      = (const float*)d_in[0];
  const float* projW  = (const float*)d_in[1];
  const float* projB  = (const float*)d_in[2];
  const float* W1     = (const float*)d_in[3];
  const float* b1     = (const float*)d_in[4];
  const float* W2     = (const float*)d_in[5];
  const float* b2     = (const float*)d_in[6];
  const float* RW1    = (const float*)d_in[7];
  const float* rb1    = (const float*)d_in[8];
  const float* RW2    = (const float*)d_in[9];
  const float* rb2    = (const float*)d_in[10];
  const float* predW  = (const float*)d_in[11];
  const float* predB  = (const float*)d_in[12];
  const int*   initT  = (const int*)d_in[13];

  float* pred = (float*)d_out;
  float* err  = pred + (size_t)N_TOK * D_INN;

  char* ws = (char*)d_ws;
  unsigned short* states = (unsigned short*)(ws + 0);       //  8,388,608 B (bf16)
  int* bucket   = (int*)(ws + 8388608);                     //    263,168 B
  int* tempers  = (int*)(ws + 8651776);                     //    262,144 B
  int* done     = (int*)(ws + 8913920);                     //    262,144 B
  int* ctrl     = (int*)(ws + 9176064);                     //      1,024 B
  const size_t wb = 9177088;
  unsigned short* projWt = (unsigned short*)(ws + wb);            // 16,384
  unsigned short* W1t    = (unsigned short*)(ws + wb + 16384);    // 294,912
  unsigned short* W2t    = (unsigned short*)(ws + wb + 311296);   // 294,912
  unsigned short* RW1t   = (unsigned short*)(ws + wb + 606208);   //   8,192
  unsigned short* RW2t   = (unsigned short*)(ws + wb + 614400);   //   2,048
  unsigned short* predWt = (unsigned short*)(ws + wb + 616448);   //  16,384

  prep_weights<<<128, 256, 0, stream>>>(projW, W1, W2, RW1, RW2, predW,
                                        projWt, W1t, W2t, RW1t, RW2t, predWt);

  proj_kernel<<<N_TOK / 16, 32, 0, stream>>>(x, projWt, projB, initT,
                                             states, tempers, done);

  for (int hop = 0; hop < N_HOPS; ++hop) {
    zero_ctrl<<<1, 64, 0, stream>>>(ctrl);
    hist_kernel<<<N_TOK / 256, 256, 0, stream>>>(tempers, done, ctrl);
    offsets_kernel<<<1, 32, 0, stream>>>(ctrl, bucket);
    scatter_kernel<<<N_TOK / 256, 256, 0, stream>>>(tempers, done, ctrl, bucket);
    mlp_kernel<<<N_TOK / 16 + T_TEMP, 32, 0, stream>>>(
        states, tempers, done, ctrl, bucket,
        W1t, W2t, b1, b2, RW1t, rb1, RW2t, rb2, hop);
  }

  pred_kernel<<<N_TOK / 16, 32, 0, stream>>>(states, predWt, predB, pred);
  err_kernel<<<(N_TOK * 32) / 256, 256, 0, stream>>>(pred, x, err);
}
